// CausalSelfAttention_37864431682228
// MI455X (gfx1250) — matrix operand order repaired and checked by reading, not compiled, not run on hardware
//
#include <hip/hip_runtime.h>
#include <hip/hip_bf16.h>

// ---------------------------------------------------------------------------
// CDNA5 (gfx1250) causal self-attention:
//   qkv = x @ qkv_w^T           (bf16 WMMA, f32 acc, TDM-staged LDS tiles)
//   q,k = rotary(rmsnorm(q,k)); v = l0*v + l1*ve     (wave32 per (t,head))
//   y   = softmax(mask(q k^T * s)) v                 (flash, WMMA)
//   out = y @ c_proj_w^T        (bf16 WMMA, f32 acc, TDM-staged LDS tiles)
// ---------------------------------------------------------------------------

typedef __attribute__((ext_vector_type(16))) __bf16 v16bf;
typedef __attribute__((ext_vector_type(8)))  __bf16 v8bf;
typedef __attribute__((ext_vector_type(8)))  float  v8f;
typedef __attribute__((ext_vector_type(4)))  unsigned int v4u;
typedef __attribute__((ext_vector_type(8)))  int    v8i;
typedef __attribute__((ext_vector_type(4)))  int    v4i;

#define DIM       1024
#define NUM_HEADS 32
#define HEAD_DIM  32
#define SEQ_T     2048
#define ATT_SCALE 0.12f
#define RMS_EPS   1.1920929e-07f
#define NEG_BIG   (-3.0e38f)

__device__ inline v16bf mk16(v8bf lo, v8bf hi) {
  v16bf r;
#pragma unroll
  for (int i = 0; i < 8; ++i) { r[i] = lo[i]; r[i + 8] = hi[i]; }
  return r;
}

__device__ inline v8f wmma_bf16(v16bf a, v16bf b, v8f c) {
  return __builtin_amdgcn_wmma_f32_16x16x32_bf16(
      false, a, false, b, (short)0, c, false, false);
}

// ---------------------------------------------------------------------------
// TDM: issue a 2D tile load (tile_h rows x tile_w dwords, row stride
// stride_dw dwords, 4-byte elements) from global into LDS at lds_off.
// D# built per CDNA5 ISA 8.3/8.4 (group0: count/lds/global/type,
// group1: data_size, dims, tile dims, strides). Groups 2/3 zero (2D).
// This toolchain exposes the 6-arg builtin: (g0, g1, g2, g3, g4, cpol).
// ---------------------------------------------------------------------------
__device__ inline void tdm_load_2d(unsigned int lds_off,
                                   unsigned long long gaddr,
                                   unsigned int tile_w_dw,
                                   unsigned int tile_h,
                                   unsigned int stride_dw) {
  v4u g0;
  g0[0] = 1u;                                   // count=1, user descriptor
  g0[1] = lds_off;                              // lds_addr (bytes)
  g0[2] = (unsigned int)gaddr;                  // global_addr[31:0]
  g0[3] = (unsigned int)((gaddr >> 32) & 0x01FFFFFFu) | (2u << 30); // [56:32]|type=2

  v8i g1;
  g1[0] = (int)(2u << 16);                      // workgroup_mask=0, data_size=4B
  g1[1] = (int)((stride_dw & 0xFFFFu) << 16);   // tensor_dim0[15:0] (use stride)
  g1[2] = (int)(((stride_dw >> 16) & 0xFFFFu) | ((tile_h & 0xFFFFu) << 16)); // d0 hi | tensor_dim1 lo
  g1[3] = (int)(((tile_h >> 16) & 0xFFFFu) | ((tile_w_dw & 0xFFFFu) << 16)); // d1 hi | tile_dim0
  g1[4] = (int)(tile_h & 0xFFFFu);              // tile_dim1 | tile_dim2=0
  g1[5] = (int)stride_dw;                       // tensor_dim0_stride[31:0]
  g1[6] = 0;                                    // stride hi | dim1_stride lo
  g1[7] = 0;

  v4i z4 = {};
  v8i z8 = {};
  __builtin_amdgcn_tensor_load_to_lds(g0, g1, z4, z4, z8, 0);
}

// ---------------------------------------------------------------------------
// Kernel 1: fp32 -> bf16 conversion
// ---------------------------------------------------------------------------
__global__ void f32_to_bf16_kernel(const float* __restrict__ in,
                                   __bf16* __restrict__ out, int n) {
  int i = blockIdx.x * blockDim.x + threadIdx.x;
  if (i < n) out[i] = (__bf16)in[i];
}

// ---------------------------------------------------------------------------
// Kernel 2: C[M x N] = A[M x K] * B[N x K]^T   (A,B bf16 K-major, C fp32)
// Block: 256 threads (8 waves), C tile 128(M) x 128(N).
// Wave (wm = wv&3, wn = wv>>2) computes 32x64 via 2x4 WMMA sub-tiles.
// K loop, step 32: A/B 128x32 bf16 slices double-buffered in LDS, filled by
// the Tensor Data Mover (one issuing wave), s_wait_tensorcnt + barriers.
// ---------------------------------------------------------------------------
__global__ void gemm_bf16_kt(const __bf16* __restrict__ A,
                             const __bf16* __restrict__ B,
                             float* __restrict__ C, int M, int N, int K) {
  __shared__ __attribute__((aligned(32))) __bf16 shA[2][128 * 32];
  __shared__ __attribute__((aligned(32))) __bf16 shB[2][128 * 32];

  const int lane = threadIdx.x & 31;
  const int wv   = threadIdx.x >> 5;
  const int col  = lane & 15;
  const int hi   = lane >> 4;
  const int wm   = wv & 3;             // 0..3  -> 32-row band
  const int wn   = wv >> 2;            // 0..1  -> 64-col band
  const int mBlk = blockIdx.y * 128;
  const int nBlk = blockIdx.x * 128;

  const unsigned long long aBase =
      (unsigned long long)(size_t)(A + (size_t)mBlk * K);
  const unsigned long long bBase =
      (unsigned long long)(size_t)(B + (size_t)nBlk * K);
  const unsigned int strideDw = (unsigned int)(K >> 1);   // dwords per row

  const int nk = K / 32;

  // Prologue: stage K-slice 0 into buffer 0.
  if (threadIdx.x == 0) {
    tdm_load_2d((unsigned int)(size_t)&shA[0][0], aBase, 16u, 128u, strideDw);
    tdm_load_2d((unsigned int)(size_t)&shB[0][0], bBase, 16u, 128u, strideDw);
  }

  v8f acc[2][4];
#pragma unroll
  for (int mi = 0; mi < 2; ++mi)
#pragma unroll
    for (int nj = 0; nj < 4; ++nj) acc[mi][nj] = (v8f){};

  for (int i = 0; i < nk; ++i) {
    const int buf = i & 1;
    if (threadIdx.x == 0) __builtin_amdgcn_s_wait_tensorcnt((short)0);
    __syncthreads();
    if (threadIdx.x == 0 && (i + 1) < nk) {
      const unsigned long long kOff = (unsigned long long)(i + 1) * 64ull; // 32 bf16
      tdm_load_2d((unsigned int)(size_t)&shA[(i + 1) & 1][0], aBase + kOff,
                  16u, 128u, strideDw);
      tdm_load_2d((unsigned int)(size_t)&shB[(i + 1) & 1][0], bBase + kOff,
                  16u, 128u, strideDw);
    }

    // A fragments for this wave: rows wm*32 + mi*16 + (lane&15)
    v16bf af[2];
#pragma unroll
    for (int mi = 0; mi < 2; ++mi) {
      const __bf16* ar = &shA[buf][(wm * 32 + mi * 16 + col) * 32];
      af[mi] = *(const v16bf*)(ar + hi * 16);
    }
#pragma unroll
    for (int nj = 0; nj < 4; ++nj) {
      const __bf16* br = &shB[buf][(wn * 64 + nj * 16 + col) * 32];
      v16bf bf = *(const v16bf*)(br + hi * 16);
      acc[0][nj] = wmma_bf16(af[0], bf, acc[0][nj]);
      acc[1][nj] = wmma_bf16(af[1], bf, acc[1][nj]);
    }
    __syncthreads();
  }

#pragma unroll
  for (int mi = 0; mi < 2; ++mi)
#pragma unroll
    for (int nj = 0; nj < 4; ++nj)
#pragma unroll
      for (int r = 0; r < 8; ++r)
        C[(size_t)(mBlk + wm * 32 + mi * 16 + r + 8 * hi) * N +
          nBlk + wn * 64 + nj * 16 + col] = acc[mi][nj][r];
}

// ---------------------------------------------------------------------------
// Kernel 3: per-(t,head) post-processing. One wave32 per (t,h); lane = dim.
// RMS norm + rotary for q,k; lambda blend for v.
// Qb,Kb: [t][h][hd] bf16.  Vt: [h][hd][t] bf16 (transposed for PV B-frags).
// ---------------------------------------------------------------------------
__global__ void qkv_postproc(const float* __restrict__ QKV,
                             const float* __restrict__ ve,
                             const float* __restrict__ lam,
                             __bf16* __restrict__ Qb,
                             __bf16* __restrict__ Kb,
                             __bf16* __restrict__ Vt) {
  const int w = blockIdx.x * (blockDim.x >> 5) + (threadIdx.x >> 5);
  const int t = w >> 5;
  const int h = w & 31;
  const int d = threadIdx.x & 31;

  const float* base = QKV + (size_t)t * (3 * DIM);
  float q = base[h * 32 + d];
  float k = base[(32 + h) * 32 + d];
  float v = base[(64 + h) * 32 + d];

  float q2 = q * q, k2 = k * k;
#pragma unroll
  for (int off = 16; off >= 1; off >>= 1) {
    q2 += __shfl_xor(q2, off, 32);
    k2 += __shfl_xor(k2, off, 32);
  }
  q *= rsqrtf(q2 * (1.0f / 32.0f) + RMS_EPS);
  k *= rsqrtf(k2 * (1.0f / 32.0f) + RMS_EPS);

  const int j = d & 15;
  float af = (j < 8) ? __powf(1.0f / 128.0f, (float)j * (1.0f / 7.0f)) : 0.0f;
  float th = (float)t * af;
  float c = __cosf(th), s = __sinf(th);
  float qp = __shfl_xor(q, 16, 32);
  float kp = __shfl_xor(k, 16, 32);
  float ssgn = (d < 16) ? s : -s;
  q = q * c + qp * ssgn;
  k = k * c + kp * ssgn;

  v = lam[0] * v + lam[1] * ve[(size_t)t * DIM + h * 32 + d];

  Qb[(size_t)t * DIM + h * 32 + d] = (__bf16)q;
  Kb[(size_t)t * DIM + h * 32 + d] = (__bf16)k;
  Vt[(size_t)h * HEAD_DIM * SEQ_T + (size_t)d * SEQ_T + t] = (__bf16)v;
}

// ---------------------------------------------------------------------------
// Kernel 4: flash attention, one wave per (head, 16-query tile), 32 keys/iter.
// ---------------------------------------------------------------------------
__global__ void attn_kernel(const __bf16* __restrict__ Qb,
                            const __bf16* __restrict__ Kb,
                            const __bf16* __restrict__ Vt,
                            __bf16* __restrict__ Yb) {
  __shared__ __attribute__((aligned(32))) __bf16 psh[4][16 * 32];

  const int wv   = threadIdx.x >> 5;
  const int lane = threadIdx.x & 31;
  const int w    = blockIdx.x * 4 + wv;
  const int head = w >> 7;
  const int t0   = (w & 127) * 16;
  const int col  = lane & 15;
  const int hi   = lane >> 4;

  const __bf16* qrow = Qb + (size_t)(t0 + col) * DIM + head * HEAD_DIM;
  v16bf qf = *(const v16bf*)(qrow + hi * 16);

  float m[8], l[8];
  v8f o1 = {}, o2 = {};
#pragma unroll
  for (int r = 0; r < 8; ++r) { m[r] = NEG_BIG; l[r] = 0.0f; }

  const __bf16* vbase = Vt + (size_t)head * HEAD_DIM * SEQ_T;

  for (int k0 = 0; k0 < t0 + 16; k0 += 32) {
    v16bf kf0 = *(const v16bf*)(Kb + (size_t)(k0 + col) * DIM +
                                head * HEAD_DIM + hi * 16);
    v16bf kf1 = *(const v16bf*)(Kb + (size_t)(k0 + 16 + col) * DIM +
                                head * HEAD_DIM + hi * 16);
    v8f z = {};
    v8f s0 = wmma_bf16(qf, kf0, z);
    v8f s1 = wmma_bf16(qf, kf1, z);

    float sA[8], sB[8], mt[8], rs[8], alpha[8];
#pragma unroll
    for (int r = 0; r < 8; ++r) {
      int qr = t0 + r + 8 * hi;
      float a = s0[r] * ATT_SCALE;
      float b = s1[r] * ATT_SCALE;
      sA[r] = (k0 + col <= qr) ? a : NEG_BIG;
      sB[r] = (k0 + 16 + col <= qr) ? b : NEG_BIG;
      mt[r] = fmaxf(sA[r], sB[r]);
    }
#pragma unroll
    for (int off = 8; off >= 1; off >>= 1)
#pragma unroll
      for (int r = 0; r < 8; ++r)
        mt[r] = fmaxf(mt[r], __shfl_xor(mt[r], off, 32));
#pragma unroll
    for (int r = 0; r < 8; ++r) {
      float mn = fmaxf(m[r], mt[r]);
      alpha[r] = __expf(m[r] - mn);
      m[r] = mn;
      sA[r] = __expf(sA[r] - mn);
      sB[r] = __expf(sB[r] - mn);
      rs[r] = sA[r] + sB[r];
    }
#pragma unroll
    for (int off = 8; off >= 1; off >>= 1)
#pragma unroll
      for (int r = 0; r < 8; ++r)
        rs[r] += __shfl_xor(rs[r], off, 32);
#pragma unroll
    for (int r = 0; r < 8; ++r) {
      l[r] = l[r] * alpha[r] + rs[r];
      o1[r] *= alpha[r];
      o2[r] *= alpha[r];
    }

    __bf16* pb = &psh[wv][0];
#pragma unroll
    for (int r = 0; r < 8; ++r) {
      int row = r + 8 * hi;
      pb[row * 32 + col]      = (__bf16)sA[r];
      pb[row * 32 + 16 + col] = (__bf16)sB[r];
    }
    asm volatile("s_wait_dscnt 0x0" ::: "memory");
    const __bf16* prow = pb + col * 32;
    v16bf pf = *(const v16bf*)(prow + hi * 16);

    v16bf vf0 = *(const v16bf*)(vbase + (size_t)col * SEQ_T + k0 + hi * 16);
    v16bf vf1 = *(const v16bf*)(vbase + (size_t)(16 + col) * SEQ_T + k0 + hi * 16);
    o1 = wmma_bf16(pf, vf0, o1);
    o2 = wmma_bf16(pf, vf1, o2);
  }

#pragma unroll
  for (int r = 0; r < 8; ++r) {
    float inv = 1.0f / l[r];
    size_t row = (size_t)(t0 + r + 8 * hi) * DIM + head * HEAD_DIM;
    Yb[row + col]      = (__bf16)(o1[r] * inv);
    Yb[row + 16 + col] = (__bf16)(o2[r] * inv);
  }
}

// ---------------------------------------------------------------------------
// Launcher. Workspace: Xb[0,4M) Wb[4M,10M) QKV f32[10M,34M) Qb[34M,38M)
// Kb[38M,42M) Vt[42M,46M) Yb[46M,50M) Cpb[50M,52M)
// ---------------------------------------------------------------------------
extern "C" void kernel_launch(void* const* d_in, const int* in_sizes, int n_in,
                              void* d_out, int out_size, void* d_ws, size_t ws_size,
                              hipStream_t stream) {
  const float* x       = (const float*)d_in[0];
  const float* ve      = (const float*)d_in[1];
  const float* qkv_w   = (const float*)d_in[2];
  const float* lambdas = (const float*)d_in[3];
  const float* c_projw = (const float*)d_in[4];
  float* out = (float*)d_out;

  const size_t MB = 1024ull * 1024ull;
  char* ws = (char*)d_ws;
  __bf16* Xb  = (__bf16*)(ws + 0);
  __bf16* Wb  = (__bf16*)(ws + 4 * MB);
  float*  QKV = (float*)(ws + 10 * MB);
  __bf16* Qb  = (__bf16*)(ws + 34 * MB);
  __bf16* Kb  = (__bf16*)(ws + 38 * MB);
  __bf16* Vt  = (__bf16*)(ws + 42 * MB);
  __bf16* Yb  = (__bf16*)(ws + 46 * MB);
  __bf16* Cpb = (__bf16*)(ws + 50 * MB);

  const int nX = SEQ_T * DIM;
  const int nW = 3 * DIM * DIM;
  const int nC = DIM * DIM;

  f32_to_bf16_kernel<<<(nX + 255) / 256, 256, 0, stream>>>(x, Xb, nX);
  f32_to_bf16_kernel<<<(nW + 255) / 256, 256, 0, stream>>>(qkv_w, Wb, nW);
  f32_to_bf16_kernel<<<(nC + 255) / 256, 256, 0, stream>>>(c_projw, Cpb, nC);

  // QKV projection: [2048 x 1024] x [3072 x 1024]^T -> [2048 x 3072]
  gemm_bf16_kt<<<dim3(3072 / 128, 2048 / 128), 256, 0, stream>>>(
      Xb, Wb, QKV, SEQ_T, 3 * DIM, DIM);

  qkv_postproc<<<(SEQ_T * NUM_HEADS) / 8, 256, 0, stream>>>(
      QKV, ve, lambdas, Qb, Kb, Vt);

  attn_kernel<<<(NUM_HEADS * (SEQ_T / 16)) / 4, 128, 0, stream>>>(
      Qb, Kb, Vt, Yb);

  // Output projection: [2048 x 1024] x [1024 x 1024]^T -> out
  gemm_bf16_kt<<<dim3(1024 / 128, 2048 / 128), 256, 0, stream>>>(
      Yb, Cpb, out, SEQ_T, DIM, DIM);
}
